// UserMultiheadAttention_24215025614966
// MI455X (gfx1250) — matrix operand order repaired and checked by reading, not compiled, not run on hardware
//
#include <hip/hip_runtime.h>
#include <hip/hip_bf16.h>

// UserMultiheadAttention for MI455X (gfx1250, wave32, WMMA).
//
// Pipeline (all GEMMs via v_wmma_f32_16x16x32_bf16, f32 accumulation):
//   K1 qkv_proj:  qb = bf16(q1 wq^T + bq), kb = ..., vbT = transpose(bf16(v1 wv^T + bv))
//   K2 scores:    S  = qb kb^T * scale + mask   (128x128 LDS-tiled, fp32 into attn region)
//   K3 softmax:   attn = softmax_row(S) + val   (fp32 to d_out; bf16 copy Pb to ws)
//   K4 attn_v:    avb = bf16(Pb @ V)            (128x128 LDS-tiled NT vs vbT)
//   K5 out_proj:  out = avb wo^T + bo           (fp32 to d_out)
//
// Workspace: qb(4MB) kb(4MB) vbT(4MB) Pb(32MB) avb(4MB) = 48 MB required.

#define L_DIM 4096
#define E_DIM 512

typedef __attribute__((ext_vector_type(16))) __bf16 v16bf;
typedef __attribute__((ext_vector_type(8)))  __bf16 v8bf;
typedef __attribute__((ext_vector_type(8)))  float  v8f;
typedef __attribute__((ext_vector_type(4)))  int    v4i;

// Pointer-to-v4i in global (AS1) / LDS (AS3) address spaces, as required by the
// gfx1250 async-load builtin prototype (probed from the round-2 diagnostic).
typedef __attribute__((address_space(1))) v4i* as1_v4i_p;
typedef __attribute__((address_space(3))) v4i* as3_v4i_p;

// Async global->LDS path (gfx1250): use the builtin when the toolchain has it.
#if defined(__HIP_DEVICE_COMPILE__) && __has_builtin(__builtin_amdgcn_global_load_async_to_lds_b128)
#define ASYNC_LDS 1
#else
#define ASYNC_LDS 0
#endif

__device__ __forceinline__ void wait_async_lds() {
#if defined(__HIP_DEVICE_COMPILE__)
#if __has_builtin(__builtin_amdgcn_s_wait_asynccnt)
  __builtin_amdgcn_s_wait_asynccnt(0);
#else
  asm volatile("s_wait_asynccnt 0x0" ::: "memory");
#endif
#endif
}

__device__ __forceinline__ __bf16 f2bf(float f) { return (__bf16)f; }

// ---- WMMA fragment loaders (CDNA5 ISA 7.12.2, wave32) ----
// A 16x32 bf16: lane holds row m=(lane&15); K elements are
//   [ kBase + 16*(lane>>4) + 0..15 ] (contiguous)
__device__ __forceinline__ v16bf loadA_bf16(const __bf16* M, int ld, int m, int kBase, int lane) {
  const __bf16* p = M + (size_t)m * ld + kBase + ((lane >> 4) << 4);
  v16bf r;
#pragma unroll
  for (int i = 0; i < 16; ++i) r[i] = p[i];
  return r;
}
__device__ __forceinline__ v16bf loadA_f32(const float* M, int ld, int m, int kBase, int lane) {
  const float* p = M + (size_t)m * ld + kBase + ((lane >> 4) << 4);
  v16bf r;
#pragma unroll
  for (int i = 0; i < 16; ++i) r[i] = f2bf(p[i]);
  return r;
}
// B 32x16 bf16 in NT form (row n of the second operand = column n of B):
// lane holds column n=(lane&15); K = kBase + 16*(lane>>4) + 0..15 (contiguous).
__device__ __forceinline__ v16bf loadBrow_f32(const float* M, int ld, int n, int kBase, int lane) {
  const float* p = M + (size_t)n * ld + kBase + ((lane >> 4) << 4);
  v16bf r;
#pragma unroll
  for (int i = 0; i < 16; ++i) r[i] = f2bf(p[i]);
  return r;
}

#define WMMA_BF16(a, b, c) \
  __builtin_amdgcn_wmma_f32_16x16x32_bf16(false, (a), false, (b), (short)0, (c), false, false)

// ================= 128x128 workgroup-tiled NT GEMM mainloop (bf16) =================
// Block = 256 threads = 8 waves. Wave w computes rows [mBase+16w, +16) x cols [nBase, +128).
// B panel (128 rows x 32 K, NT source) staged in LDS each K-step; shared by all 8 waves.
#define TILE_LDB 40  // row pitch in elements (128 rows * 40 * 2B = 10 KB; 16B-aligned chunks)

__device__ __forceinline__ void gemm_tile_128x128(
    const __bf16* __restrict__ A, int lda, int mBase,
    const __bf16* __restrict__ B, int ldb, int nBase,
    int K, __bf16* tile, v8f (&acc)[8]) {
  const int tid  = threadIdx.x;
  const int lane = tid & 31;
  const int wid  = tid >> 5;
  const int mA   = mBase + 16 * wid + (lane & 15);

  for (int k = 0; k < K; k += 32) {
    // ---- stage B[nBase..+128)[k..k+32) into LDS: 512 chunks of 16B, 2 per thread ----
#pragma unroll
    for (int i = 0; i < 2; ++i) {
      const int c   = tid + 256 * i;      // chunk id 0..511
      const int row = c >> 2;             // 4 chunks per 64B row
      const int ch  = (c & 3) << 3;       // element offset within row: 0,8,16,24
      const __bf16* g = B + (size_t)(nBase + row) * ldb + k + ch;
      __bf16* l = tile + row * TILE_LDB + ch;
#if ASYNC_LDS
      __builtin_amdgcn_global_load_async_to_lds_b128((as1_v4i_p)g, (as3_v4i_p)l, 0, 0);
#else
      *(v8bf*)l = *(const v8bf*)g;
#endif
    }
#if ASYNC_LDS
    wait_async_lds();
#endif
    __syncthreads();

    // ---- compute: 1 A fragment, 8 B fragments from LDS, 8 WMMAs ----
    const v16bf a = loadA_bf16(A, lda, mA, k, lane);
    const int brow0 = lane & 15;
    const int koff  = (lane >> 4) << 4;
#pragma unroll
    for (int j = 0; j < 8; ++j) {
      const __bf16* p = tile + (16 * j + brow0) * TILE_LDB + koff;
      const v8bf lo = *(const v8bf*)p;          // 16B-aligned LDS loads
      const v8bf hi = *(const v8bf*)(p + 8);
      const v16bf b = __builtin_shufflevector(lo, hi, 0, 1, 2, 3, 4, 5, 6, 7,
                                              8, 9, 10, 11, 12, 13, 14, 15);
      acc[j] = WMMA_BF16(a, b, acc[j]);
    }
    __syncthreads();  // protect LDS panel before next stage overwrites it
  }
}

// ================= K1: fused Q/K/V projection (f32 in -> bf16 out) =================
__global__ __launch_bounds__(256) void qkv_proj_kernel(
    const float* __restrict__ q1, const float* __restrict__ k1, const float* __restrict__ v1,
    const float* __restrict__ wq, const float* __restrict__ wk, const float* __restrict__ wv,
    const float* __restrict__ bq, const float* __restrict__ bk, const float* __restrict__ bv,
    __bf16* __restrict__ qb, __bf16* __restrict__ kb, __bf16* __restrict__ vbT) {
  const int mat = blockIdx.y;  // 0=q, 1=k, 2=v
  const float* A    = (mat == 0) ? q1 : (mat == 1) ? k1 : v1;
  const float* W    = (mat == 0) ? wq : (mat == 1) ? wk : wv;
  const float* bias = (mat == 0) ? bq : (mat == 1) ? bk : bv;

  const int lane  = threadIdx.x & 31;
  const int gw    = (blockIdx.x * blockDim.x + threadIdx.x) >> 5;
  const int nGrp  = E_DIM / 64;  // 8
  const int mBase = (gw / nGrp) * 16;
  const int nBase = (gw % nGrp) * 64;
  const int mA    = mBase + (lane & 15);

  v8f acc[4] = {};
  for (int k = 0; k < E_DIM; k += 32) {
    v16bf a = loadA_f32(A, E_DIM, mA, k, lane);
#pragma unroll
    for (int j = 0; j < 4; ++j) {
      v16bf b = loadBrow_f32(W, E_DIM, nBase + 16 * j + (lane & 15), k, lane);
      acc[j] = WMMA_BF16(a, b, acc[j]);
    }
  }
  const int n0 = lane & 15;
  const int mo = (lane >> 4) << 3;
#pragma unroll
  for (int j = 0; j < 4; ++j) {
    const int n = nBase + 16 * j + n0;
    const float bi = bias[n];
#pragma unroll
    for (int r = 0; r < 8; ++r) {
      const int m = mBase + mo + r;
      const float val = acc[j][r] + bi;
      if (mat == 2) vbT[(size_t)n * L_DIM + m] = f2bf(val);  // store V transposed [E,L]
      else if (mat == 0) qb[(size_t)m * E_DIM + n] = f2bf(val);
      else kb[(size_t)m * E_DIM + n] = f2bf(val);
    }
  }
}

// ================= K2: S = qb kb^T * scale + mask (fp32, into attn region) =================
__global__ __launch_bounds__(256) void scores_kernel(
    const __bf16* __restrict__ qb, const __bf16* __restrict__ kb,
    const float* __restrict__ mask, float* __restrict__ S) {
  __shared__ __bf16 tile[128 * TILE_LDB];
  const int mBase = blockIdx.y * 128;
  const int nBase = blockIdx.x * 128;
  const float scale = 0.04419417382415922f;  // 1/sqrt(512)

  v8f acc[8] = {};
  gemm_tile_128x128(qb, E_DIM, mBase, kb, E_DIM, nBase, E_DIM, tile, acc);

  const int lane = threadIdx.x & 31;
  const int wid  = threadIdx.x >> 5;
  const int n0   = lane & 15;
  const int mo   = (lane >> 4) << 3;
#pragma unroll
  for (int j = 0; j < 8; ++j) {
    const int n = nBase + 16 * j + n0;
#pragma unroll
    for (int r = 0; r < 8; ++r) {
      const size_t idx = (size_t)(mBase + 16 * wid + mo + r) * L_DIM + n;
      S[idx] = acc[j][r] * scale + mask[idx];
    }
  }
}

// ================= K3: row softmax in place, + pref_val, bf16 copy =================
__global__ __launch_bounds__(256) void softmax_kernel(
    const float* __restrict__ pval, float* __restrict__ attn, __bf16* __restrict__ Pb) {
  const int row = blockIdx.x;
  const int tid = threadIdx.x;
  float* srow = attn + (size_t)row * L_DIM;

  float x[16];
  float m = -3.402823466e38f;
#pragma unroll
  for (int i = 0; i < 16; ++i) { x[i] = srow[tid + i * 256]; m = fmaxf(m, x[i]); }

  __shared__ float redmax[8];
  __shared__ float redsum[8];
  const int wid = tid >> 5, lane = tid & 31;

#pragma unroll
  for (int off = 16; off > 0; off >>= 1) m = fmaxf(m, __shfl_xor(m, off, 32));
  if (lane == 0) redmax[wid] = m;
  __syncthreads();
  m = redmax[0];
#pragma unroll
  for (int i = 1; i < 8; ++i) m = fmaxf(m, redmax[i]);

  float e[16];
  float s = 0.0f;
#pragma unroll
  for (int i = 0; i < 16; ++i) { e[i] = __expf(x[i] - m); s += e[i]; }
#pragma unroll
  for (int off = 16; off > 0; off >>= 1) s += __shfl_xor(s, off, 32);
  if (lane == 0) redsum[wid] = s;
  __syncthreads();
  s = 0.0f;
#pragma unroll
  for (int i = 0; i < 8; ++i) s += redsum[i];
  const float inv = 1.0f / s;

  const float* vrow = pval + (size_t)row * L_DIM;
  __bf16* prow = Pb + (size_t)row * L_DIM;
#pragma unroll
  for (int i = 0; i < 16; ++i) {
    const int c = tid + i * 256;
    const float y = e[i] * inv + vrow[c];
    srow[c] = y;          // fp32 attn output
    prow[c] = f2bf(y);    // bf16 operand for attn @ V
  }
}

// ================= K4: AV = Pb @ V  (NT vs vbT), bf16 out =================
__global__ __launch_bounds__(256) void attnv_kernel(
    const __bf16* __restrict__ Pb, const __bf16* __restrict__ vbT, __bf16* __restrict__ avb) {
  __shared__ __bf16 tile[128 * TILE_LDB];
  const int mBase = blockIdx.y * 128;
  const int nBase = blockIdx.x * 128;

  v8f acc[8] = {};
  gemm_tile_128x128(Pb, L_DIM, mBase, vbT, L_DIM, nBase, L_DIM, tile, acc);

  const int lane = threadIdx.x & 31;
  const int wid  = threadIdx.x >> 5;
  const int n0   = lane & 15;
  const int mo   = (lane >> 4) << 3;
#pragma unroll
  for (int j = 0; j < 8; ++j) {
    const int n = nBase + 16 * j + n0;
#pragma unroll
    for (int r = 0; r < 8; ++r)
      avb[(size_t)(mBase + 16 * wid + mo + r) * E_DIM + n] = f2bf(acc[j][r]);
  }
}

// ================= K5: out = avb wo^T + bo (fp32 out) =================
__global__ __launch_bounds__(256) void outproj_kernel(
    const __bf16* __restrict__ avb, const float* __restrict__ wo,
    const float* __restrict__ bo, float* __restrict__ out) {
  const int lane  = threadIdx.x & 31;
  const int gw    = (blockIdx.x * blockDim.x + threadIdx.x) >> 5;
  const int nGrp  = E_DIM / 64;  // 8
  const int mBase = (gw / nGrp) * 16;
  const int nBase = (gw % nGrp) * 64;
  const int mA    = mBase + (lane & 15);

  v8f acc[4] = {};
  for (int k = 0; k < E_DIM; k += 32) {
    v16bf a = loadA_bf16(avb, E_DIM, mA, k, lane);
#pragma unroll
    for (int j = 0; j < 4; ++j) {
      v16bf b = loadBrow_f32(wo, E_DIM, nBase + 16 * j + (lane & 15), k, lane);
      acc[j] = WMMA_BF16(a, b, acc[j]);
    }
  }
  const int n0 = lane & 15;
  const int mo = (lane >> 4) << 3;
#pragma unroll
  for (int j = 0; j < 4; ++j) {
    const int n = nBase + 16 * j + n0;
    const float bi = bo[n];
#pragma unroll
    for (int r = 0; r < 8; ++r)
      out[(size_t)(mBase + mo + r) * E_DIM + n] = acc[j][r] + bi;
  }
}

extern "C" void kernel_launch(void* const* d_in, const int* in_sizes, int n_in,
                              void* d_out, int out_size, void* d_ws, size_t ws_size,
                              hipStream_t stream) {
  const float* q1    = (const float*)d_in[0];
  const float* k1    = (const float*)d_in[1];
  const float* v1    = (const float*)d_in[2];
  const float* pval  = (const float*)d_in[3];
  const float* pmask = (const float*)d_in[4];
  const float* wq    = (const float*)d_in[5];
  const float* bq    = (const float*)d_in[6];
  const float* wk    = (const float*)d_in[7];
  const float* bk    = (const float*)d_in[8];
  const float* wv    = (const float*)d_in[9];
  const float* bv    = (const float*)d_in[10];
  const float* wo    = (const float*)d_in[11];
  const float* bo    = (const float*)d_in[12];

  float* out  = (float*)d_out;                 // [L,E]
  float* attn = out + (size_t)L_DIM * E_DIM;   // [L,L]

  __bf16* qb  = (__bf16*)d_ws;
  __bf16* kb  = qb  + (size_t)L_DIM * E_DIM;
  __bf16* vbT = kb  + (size_t)L_DIM * E_DIM;   // [E,L]
  __bf16* Pb  = vbT + (size_t)L_DIM * E_DIM;   // [L,L]
  __bf16* avb = Pb  + (size_t)L_DIM * L_DIM;   // [L,E]
  (void)in_sizes; (void)n_in; (void)out_size; (void)ws_size;

  // K1: (L/16)*(E/64) = 2048 waves per matrix -> 256 blocks, y = matrix
  qkv_proj_kernel<<<dim3(256, 3), 256, 0, stream>>>(q1, k1, v1, wq, wk, wv, bq, bk, bv,
                                                    qb, kb, vbT);
  // K2: 128x128 tiles -> (32, 32) blocks of 256 threads
  scores_kernel<<<dim3(L_DIM / 128, L_DIM / 128), 256, 0, stream>>>(qb, kb, pmask, attn);
  // K3: one block per row
  softmax_kernel<<<L_DIM, 256, 0, stream>>>(pval, attn, Pb);
  // K4: 128x128 tiles -> (4, 32) blocks
  attnv_kernel<<<dim3(E_DIM / 128, L_DIM / 128), 256, 0, stream>>>(Pb, vbT, avb);
  // K5: 2048 waves -> 256 blocks
  outproj_kernel<<<256, 256, 0, stream>>>(avb, wo, bo, out);
}
